// MaskedMultiHeadAttention_81595788690053
// MI455X (gfx1250) — matrix operand order repaired and checked by reading, not compiled, not run on hardware
//
#include <hip/hip_runtime.h>
#include <hip/hip_bf16.h>

typedef unsigned short u16;
typedef __attribute__((ext_vector_type(16))) __bf16 v16bf;
typedef __attribute__((ext_vector_type(8)))  float  v8f;

union U4H { uint4 v; u16 h[8]; };
union V16U { v16bf v; uint4 u[2]; };

__device__ __forceinline__ u16 f2bf(float f) {
    unsigned u = __float_as_uint(f);
    u += 0x7FFFu + ((u >> 16) & 1u);   // round-to-nearest-even
    return (u16)(u >> 16);
}

__device__ __forceinline__ v16bf mk16(uint4 lo, uint4 hi) {
    V16U t; t.u[0] = lo; t.u[1] = hi; return t.v;
}

__device__ __forceinline__ v8f zero8() {
    v8f z;
#pragma unroll
    for (int i = 0; i < 8; ++i) z[i] = 0.0f;
    return z;
}

__device__ __forceinline__ v8f wmma_bf16(v16bf a, v16bf b, v8f c) {
    return __builtin_amdgcn_wmma_f32_16x16x32_bf16(false, a, false, b, (short)0, c, false, false);
}

// gfx1250 async global->LDS copy (bypasses VGPRs, tracked by ASYNCcnt).
// lds_off = wave-relative LDS byte address (low 32 bits of a generic LDS pointer).
__device__ __forceinline__ void async_load_b128(unsigned lds_off, const void* g) {
    asm volatile("global_load_async_to_lds_b128 %0, %1, off"
                 :: "v"(lds_off), "v"(g) : "memory");
}
__device__ __forceinline__ void wait_async0() {
    asm volatile("s_wait_asynccnt 0x0" ::: "memory");
}

// ---------------------------------------------------------------- cast kernel
__global__ void cast_f32_bf16(const float* __restrict__ src, u16* __restrict__ dst, int n) {
    int i = blockIdx.x * blockDim.x + threadIdx.x;
    if (i < n) dst[i] = f2bf(src[i]);
}

// ---------------------------------------------------------------- GEMM (bf16 -> f32 acc)
// A: [M,K] bf16 row-major.  Bm: [K,N] bf16 row-major.
// MODE 0: qkv epilogue (scatter q/k/vT, q pre-scaled by 0.125)
// MODE 1: plain f32 store to outF [M,N]
template <int MODE>
__global__ __launch_bounds__(256) void gemm_bf16(
    const u16* __restrict__ A, const u16* __restrict__ Bm,
    u16* __restrict__ qws, u16* __restrict__ kws, u16* __restrict__ vwsT,
    float* __restrict__ outF, int M, int N, int K)
{
    __shared__ u16 As[128][40];   // [m][k], +8 pad
    __shared__ u16 Bs[128][40];   // [n][k] transposed, +8 pad

    const int tid  = threadIdx.x;
    const int lane = tid & 31, wid = tid >> 5;
    const int l16  = lane & 15, hi = lane >> 4;
    const int wm   = (wid >> 2) * 64;     // wave M offset in block tile
    const int wn   = (wid & 3) * 32;      // wave N offset in block tile
    const int m0   = blockIdx.y * 128, n0 = blockIdx.x * 128;

    v8f acc[4][2];
#pragma unroll
    for (int i = 0; i < 4; ++i)
#pragma unroll
        for (int j = 0; j < 2; ++j) acc[i][j] = zero8();

    for (int ks = 0; ks < K; ks += 32) {
        __syncthreads();
        // fill As via async DMA: 128x32 halves = 512 chunks of 8
#pragma unroll
        for (int i = 0; i < 2; ++i) {
            int c = tid * 2 + i;
            int row = c >> 2, col = (c & 3) * 8;
            async_load_b128((unsigned)(size_t)&As[row][col],
                            &A[(size_t)(m0 + row) * K + ks + col]);
        }
        // fill Bs transposed: read [k][n] coalesced, write [n][k]
#pragma unroll
        for (int i = 0; i < 2; ++i) {
            int c = tid * 2 + i;
            int kk = c >> 4, nn = (c & 15) * 8;
            U4H t; t.v = *(const uint4*)&Bm[(size_t)(ks + kk) * N + n0 + nn];
#pragma unroll
            for (int j = 0; j < 8; ++j) Bs[nn + j][kk] = t.h[j];
        }
        wait_async0();
        __syncthreads();

        const int kb = hi * 8;    // A frag K base per half-wave
        const int kg = hi * 8;    // B frag K base per half-wave
        v16bf afr[4], bfr[2];
#pragma unroll
        for (int i = 0; i < 4; ++i) {
            int row = wm + i * 16 + l16;
            afr[i] = mk16(*(const uint4*)&As[row][kb],
                          *(const uint4*)&As[row][16 + kb]);
        }
#pragma unroll
        for (int j = 0; j < 2; ++j) {
            int nrow = wn + j * 16 + l16;
            bfr[j] = mk16(*(const uint4*)&Bs[nrow][kg],
                          *(const uint4*)&Bs[nrow][kg + 16]);
        }
#pragma unroll
        for (int i = 0; i < 4; ++i)
#pragma unroll
            for (int j = 0; j < 2; ++j)
                acc[i][j] = wmma_bf16(afr[i], bfr[j], acc[i][j]);
    }

    // epilogue
#pragma unroll
    for (int i = 0; i < 4; ++i)
#pragma unroll
        for (int j = 0; j < 2; ++j)
#pragma unroll
            for (int r = 0; r < 8; ++r) {
                int m = m0 + wm + i * 16 + r + 8 * hi;
                int n = n0 + wn + j * 16 + l16;
                float val = acc[i][j][r];
                if (MODE == 0) {
                    int b = m >> 11, t = m & 2047;          // T=2048
                    int sect = n >> 10;                      // D=1024
                    int hh = (n & 1023) >> 6, hd = n & 63;   // HD=64
                    size_t bh = (size_t)(b * 16 + hh);
                    if (sect == 0)      qws[(bh * 2048 + t) * 64 + hd] = f2bf(val * 0.125f);
                    else if (sect == 1) kws[(bh * 2048 + t) * 64 + hd] = f2bf(val);
                    else                vwsT[(bh * 64 + hd) * 2048 + t] = f2bf(val);
                } else {
                    outF[(size_t)m * N + n] = val;
                }
            }
}

// ---------------------------------------------------------------- flash attention
// qws/kws: [B*H][T][64] bf16 (q pre-scaled), vwsT: [B*H][64][T] bf16
// attn: [B][T][D] bf16
__global__ __launch_bounds__(128) void flash_attn(
    const u16* __restrict__ qws, const u16* __restrict__ kws,
    const u16* __restrict__ vwsT, u16* __restrict__ attn)
{
    constexpr int T = 2048, HD = 64, D = 1024;
    __shared__ u16 Ksh[64][72];       // [key][hd]
    __shared__ u16 Vsh[64][72];       // [hd][key]
    __shared__ u16 Psh[4][16][72];    // per-wave P strip [qrow][key]

    const int qt = blockIdx.x;        // 0..31 (64-row q tile)
    const int bh = blockIdx.y;        // 0..63
    const int tid = threadIdx.x;
    const int lane = tid & 31, w = tid >> 5;
    const int l16 = lane & 15, hi = lane >> 4;
    const int kb = hi * 8, kg = hi * 8;

    const size_t baseQK = (size_t)bh * T * HD;
    const size_t baseVT = (size_t)bh * HD * T;

    // Q fragments for this wave's 16 rows (kept in registers, reused each k-tile)
    const int qrow = qt * 64 + w * 16 + l16;
    const u16* qp = qws + baseQK + (size_t)qrow * HD;
    v16bf qf0 = mk16(*(const uint4*)&qp[kb],      *(const uint4*)&qp[kb + 16]);
    v16bf qf1 = mk16(*(const uint4*)&qp[32 + kb], *(const uint4*)&qp[48 + kb]);

    v8f accO[4];
#pragma unroll
    for (int i = 0; i < 4; ++i) accO[i] = zero8();
    float mrow[8], lrow[8];
#pragma unroll
    for (int r = 0; r < 8; ++r) { mrow[r] = -1e30f; lrow[r] = 0.0f; }

    for (int kt = 0; kt <= qt; ++kt) {
        const int t0 = kt * 64;
        __syncthreads();
        // async DMA stage: K tile [key][hd] and V tile [hd][key]
        // (512 chunks of 8 halves each, 4 per thread)
#pragma unroll
        for (int i = 0; i < 4; ++i) {
            int c = tid * 4 + i;
            int row = c >> 3, col = (c & 7) * 8;
            async_load_b128((unsigned)(size_t)&Ksh[row][col],
                            &kws[baseQK + (size_t)(t0 + row) * HD + col]);
        }
#pragma unroll
        for (int i = 0; i < 4; ++i) {
            int c = tid * 4 + i;
            int row = c >> 3, col = (c & 7) * 8;
            async_load_b128((unsigned)(size_t)&Vsh[row][col],
                            &vwsT[baseVT + (size_t)row * T + t0 + col]);
        }
        wait_async0();
        __syncthreads();

        // S = Q K^T : 4 key-subtiles, 2 WMMAs each (hd 0-31, 32-63)
        v8f s[4];
#pragma unroll
        for (int sn = 0; sn < 4; ++sn) {
            int krow = sn * 16 + l16;
            v16bf b0 = mk16(*(const uint4*)&Ksh[krow][kg],
                            *(const uint4*)&Ksh[krow][kg + 16]);
            v16bf b1 = mk16(*(const uint4*)&Ksh[krow][32 + kg],
                            *(const uint4*)&Ksh[krow][32 + kg + 16]);
            v8f c = zero8();
            c = wmma_bf16(qf0, b0, c);
            c = wmma_bf16(qf1, b1, c);
            s[sn] = c;
        }
        if (kt == qt) {   // causal mask, diagonal tile only (uniform branch)
#pragma unroll
            for (int sn = 0; sn < 4; ++sn) {
                int key = t0 + sn * 16 + l16;
#pragma unroll
                for (int r = 0; r < 8; ++r) {
                    int qg = qt * 64 + w * 16 + r + 8 * hi;
                    if (key > qg) s[sn][r] = -1e30f;
                }
            }
        }
        // online softmax: row stats via 16-lane half-wave shuffles
        float mnew[8], alpha[8];
#pragma unroll
        for (int r = 0; r < 8; ++r) {
            float m = fmaxf(fmaxf(s[0][r], s[1][r]), fmaxf(s[2][r], s[3][r]));
            m = fmaxf(m, __shfl_xor(m, 1));
            m = fmaxf(m, __shfl_xor(m, 2));
            m = fmaxf(m, __shfl_xor(m, 4));
            m = fmaxf(m, __shfl_xor(m, 8));
            mnew[r]  = fmaxf(mrow[r], m);
            alpha[r] = __expf(mrow[r] - mnew[r]);
            mrow[r]  = mnew[r];
        }
#pragma unroll
        for (int r = 0; r < 8; ++r) {
            float sum = 0.0f;
#pragma unroll
            for (int sn = 0; sn < 4; ++sn) {
                float p = __expf(s[sn][r] - mnew[r]);
                s[sn][r] = p;
                sum += p;
            }
            sum += __shfl_xor(sum, 1);
            sum += __shfl_xor(sum, 2);
            sum += __shfl_xor(sum, 4);
            sum += __shfl_xor(sum, 8);
            lrow[r] = lrow[r] * alpha[r] + sum;
#pragma unroll
            for (int i = 0; i < 4; ++i) accO[i][r] *= alpha[r];
        }
        // P: C-layout -> LDS -> A-layout (wave-private region, LDS in-order per wave)
#pragma unroll
        for (int sn = 0; sn < 4; ++sn)
#pragma unroll
            for (int r = 0; r < 8; ++r)
                Psh[w][r + 8 * hi][sn * 16 + l16] = f2bf(s[sn][r]);
        const u16* pp = &Psh[w][l16][0];
        v16bf pf0 = mk16(*(const uint4*)&pp[kb],      *(const uint4*)&pp[kb + 16]);
        v16bf pf1 = mk16(*(const uint4*)&pp[32 + kb], *(const uint4*)&pp[48 + kb]);
        // O += P V : 4 hd-subtiles, 2 WMMAs each (key 0-31, 32-63)
#pragma unroll
        for (int sn = 0; sn < 4; ++sn) {
            int vrow = sn * 16 + l16;
            v16bf b0 = mk16(*(const uint4*)&Vsh[vrow][kg],
                            *(const uint4*)&Vsh[vrow][kg + 16]);
            v16bf b1 = mk16(*(const uint4*)&Vsh[vrow][32 + kg],
                            *(const uint4*)&Vsh[vrow][32 + kg + 16]);
            accO[sn] = wmma_bf16(pf0, b0, accO[sn]);
            accO[sn] = wmma_bf16(pf1, b1, accO[sn]);
        }
    }

    // normalize + store to attn [B][T][D]
    const int b = bh >> 4, h = bh & 15;
#pragma unroll
    for (int sn = 0; sn < 4; ++sn)
#pragma unroll
        for (int r = 0; r < 8; ++r) {
            int t = qt * 64 + w * 16 + r + 8 * hi;
            int col = h * 64 + sn * 16 + l16;
            attn[((size_t)b * T + t) * D + col] = f2bf(accO[sn][r] / lrow[r]);
        }
}

// ---------------------------------------------------------------- launch
extern "C" void kernel_launch(void* const* d_in, const int* in_sizes, int n_in,
                              void* d_out, int out_size, void* d_ws, size_t ws_size,
                              hipStream_t stream)
{
    constexpr int B = 4, T = 2048, D = 1024;
    const float* x    = (const float*)d_in[0];
    const float* wqkv = (const float*)d_in[1];
    const float* wo   = (const float*)d_in[2];
    float* out = (float*)d_out;

    const size_t nx    = (size_t)B * T * D;   // 8388608
    const size_t nwqkv = (size_t)D * 3 * D;   // 3145728
    const size_t nwo   = (size_t)D * D;       // 1048576

    u16* ws    = (u16*)d_ws;
    u16* xb    = ws;
    u16* wqkvb = xb + nx;
    u16* wob   = wqkvb + nwqkv;
    u16* qws   = wob + nwo;
    u16* kws   = qws + nx;
    u16* vwsT  = kws + nx;
    u16* attnb = vwsT + nx;

    cast_f32_bf16<<<(int)((nx    + 255) / 256), 256, 0, stream>>>(x,    xb,    (int)nx);
    cast_f32_bf16<<<(int)((nwqkv + 255) / 256), 256, 0, stream>>>(wqkv, wqkvb, (int)nwqkv);
    cast_f32_bf16<<<(int)((nwo   + 255) / 256), 256, 0, stream>>>(wo,   wob,   (int)nwo);

    // QKV projection: M=8192, N=3072, K=1024
    gemm_bf16<0><<<dim3(3072 / 128, 8192 / 128), 256, 0, stream>>>(
        xb, wqkvb, qws, kws, vwsT, nullptr, 8192, 3072, 1024);

    // causal flash attention: grid (q-tiles, B*H)
    flash_attn<<<dim3(T / 64, B * 16), 128, 0, stream>>>(qws, kws, vwsT, attnb);

    // output projection: M=8192, N=1024, K=1024
    gemm_bf16<1><<<dim3(1024 / 128, 8192 / 128), 256, 0, stream>>>(
        attnb, wob, nullptr, nullptr, nullptr, out, 8192, 1024, 1024);
}
